// LTCCell_8272107012244
// MI455X (gfx1250) — matrix operand order repaired and checked by reading, not compiled, not run on hardware
//
#include <hip/hip_runtime.h>

typedef __attribute__((ext_vector_type(16))) _Float16 v16h;
typedef __attribute__((ext_vector_type(8)))  _Float16 v8h;
typedef __attribute__((ext_vector_type(8)))  float    v8f;
typedef __attribute__((ext_vector_type(4)))  float    v4f;
typedef __attribute__((ext_vector_type(4)))  int      v4i;

#define BB    128
#define TT    256
#define II    128
#define HH    512
#define NUNF  3
#define EPSF  1e-8f
#define LDST  520   // f16 LDS row stride (1040B) for the h A-fragment staging
#define IPST  520   // f32 LDS row stride for the ip tile double buffer

#if defined(__gfx1250__) && __has_builtin(__builtin_amdgcn_global_load_async_to_lds_b128) && __has_builtin(__builtin_amdgcn_s_wait_asynccnt)
#define USE_ASYNC 1
typedef __attribute__((address_space(1))) v4i g_v4i;   // global (device) AS
typedef __attribute__((address_space(3))) v4i l_v4i;   // LDS AS
#else
#define USE_ASYNC 0
#endif

// ---------------------------------------------------------------------------
// f32 -> f16 elementwise convert (for W_rec and W_in, done once per launch)
// ---------------------------------------------------------------------------
__global__ void __launch_bounds__(256) cvt_f32_f16(const float* __restrict__ s,
                                                   _Float16* __restrict__ d, int n) {
  int i = blockIdx.x * 256 + threadIdx.x;
  if (i < n) d[i] = (_Float16)s[i];
}

// ---------------------------------------------------------------------------
// Input projection: ip[r, n] = sum_i x[r, i] * W_in[n, i] + b[n];  r = b*T + t.
// One wave per 16x16 output tile, K-loop = 128/32, f16 WMMA / f32 accumulate.
// ---------------------------------------------------------------------------
__global__ void __launch_bounds__(256) input_gemm(const float* __restrict__ x,
                                                  const _Float16* __restrict__ win16,
                                                  const float* __restrict__ bias,
                                                  float* __restrict__ ip) {
  const int lane  = threadIdx.x & 31;
  const int wv    = threadIdx.x >> 5;
  const int g     = blockIdx.x * 8 + wv;     // global wave id
  const int mtile = g >> 5;                  // 0..2047
  const int ntile = g & 31;                  // 0..31
  const int nloc  = lane & 15;
  const int hi    = lane >> 4;
  const int n     = ntile * 16 + nloc;

  const float* arow = x + (mtile * 16 + nloc) * II;

  float bv = bias[n];
  v8f c = {bv, bv, bv, bv, bv, bv, bv, bv};

#pragma unroll
  for (int kt = 0; kt < II / 32; ++kt) {
    const int k0 = kt * 32 + hi * 16;
    v8f xlo = *(const v8f*)(arow + k0);
    v8f xhi = *(const v8f*)(arow + k0 + 8);
    v16h a;
#pragma unroll
    for (int e = 0; e < 8; ++e) { a[e] = (_Float16)xlo[e]; a[e + 8] = (_Float16)xhi[e]; }
    v16h b = *(const v16h*)(win16 + n * II + kt * 32 + hi * 16);
    c = __builtin_amdgcn_wmma_f32_16x16x32_f16(false, a, false, b, (short)0, c,
                                               false, false);
  }

#pragma unroll
  for (int v = 0; v < 8; ++v)
    ip[(mtile * 16 + v + hi * 8) * HH + n] = c[v];
}

// ---------------------------------------------------------------------------
// Persistent LTC scan. 8 blocks x 512 threads (16 waves); block owns 16 batch
// rows for all T steps. Each wave owns 2 n-tiles (2 independent WMMA chains of
// 16). h lives in registers (C/D layout) and is re-staged through LDS as f16
// A-fragments each unfold. The next timestep's inp_proj tile is prefetched
// into an LDS double buffer with async global->LDS DMA (ASYNCcnt).
// ---------------------------------------------------------------------------
__global__ void __launch_bounds__(512) ltc_scan(const float* __restrict__ ip,
                                                const _Float16* __restrict__ wrec,
                                                const float* __restrict__ tau,
                                                const float* __restrict__ erev,
                                                const float* __restrict__ gleak,
                                                const float* __restrict__ eleak,
                                                float* __restrict__ out) {
  __shared__ __align__(16) _Float16 hA[16 * LDST];
  __shared__ __align__(16) float    ipL[2][16 * IPST];

  const int tid  = threadIdx.x;
  const int lane = tid & 31;
  const int wave = tid >> 5;          // 0..15
  const int nloc = lane & 15;
  const int hi   = lane >> 4;
  const int b0   = blockIdx.x * 16;

  // Per-neuron parameters (lane-resident, one per owned n-tile)
  float gl[2], er[2], el[2], itau[2];
  int   ncol[2];
#pragma unroll
  for (int j = 0; j < 2; ++j) {
    const int n = (wave * 2 + j) * 16 + nloc;
    ncol[j] = n;
    gl[j]   = fmaxf(gleak[n], EPSF);
    er[j]   = erev[n];
    el[j]   = eleak[n];
    itau[j] = (1.0f / (float)NUNF) / fmaxf(tau[n], EPSF);
  }

  const v8f vzero = {0.f, 0.f, 0.f, 0.f, 0.f, 0.f, 0.f, 0.f};
  v8f h[2];
#pragma unroll
  for (int j = 0; j < 2; ++j) h[j] = vzero;

  const int arow_off = nloc * LDST;   // A-fragment row for this lane

  // ---- ip tile prefetch: 16 rows x 512 f32 = 2048 x b128, 4 chunks/thread
  auto prefetch_ip = [&](int t, int buf) {
#pragma unroll
    for (int i = 0; i < 4; ++i) {
      const int id  = tid + 512 * i;
      const int row = id >> 7;            // 128 chunks per row
      const int col = (id & 127) * 4;
      const float* g = ip + ((size_t)((b0 + row) * TT + t)) * HH + col;
      float* l = &ipL[buf][row * IPST + col];
#if USE_ASYNC
      __builtin_amdgcn_global_load_async_to_lds_b128((g_v4i*)g, (l_v4i*)l, 0, 0);
#else
      *(v4f*)l = *(const v4f*)g;
#endif
    }
  };

  prefetch_ip(0, 0);

  for (int t = 0; t < TT; ++t) {
#if USE_ASYNC
    __builtin_amdgcn_s_wait_asynccnt(0);
#endif
    __syncthreads();  // prefetched tile visible; prior buffer reads retired

    // Pull this step's ip tile into registers (reused across unfolds)
    v8f ipr[2];
#pragma unroll
    for (int j = 0; j < 2; ++j)
#pragma unroll
      for (int v = 0; v < 8; ++v)
        ipr[j][v] = ipL[t & 1][(v + hi * 8) * IPST + ncol[j]];

    if (t + 1 < TT) prefetch_ip(t + 1, (t + 1) & 1);

    for (int u = 0; u < NUNF; ++u) {
      // Stage h (f32 regs, D layout) -> LDS f16 row-major [m][k]
      __syncthreads();  // WAR vs previous unfold's A-fragment reads
#pragma unroll
      for (int j = 0; j < 2; ++j) {
        const int col = ncol[j];
#pragma unroll
        for (int v = 0; v < 8; ++v)
          hA[(v + hi * 8) * LDST + col] = (_Float16)h[j][v];
      }
      __syncthreads();

      v8f acc[2];
#pragma unroll
      for (int j = 0; j < 2; ++j) acc[j] = ipr[j];

#pragma unroll 8
      for (int kt = 0; kt < HH / 32; ++kt) {
        const int k0 = kt * 32 + hi * 16;
        v8h alo = *(const v8h*)&hA[arow_off + k0];
        v8h ahi = *(const v8h*)&hA[arow_off + k0 + 8];
        v16h a = __builtin_shufflevector(alo, ahi, 0, 1, 2, 3, 4, 5, 6, 7, 8, 9,
                                         10, 11, 12, 13, 14, 15);
#pragma unroll
        for (int j = 0; j < 2; ++j) {
          v16h b = *(const v16h*)(wrec + ncol[j] * HH + kt * 32 + hi * 16);
          acc[j] = __builtin_amdgcn_wmma_f32_16x16x32_f16(false, a, false, b,
                                                          (short)0, acc[j],
                                                          false, false);
        }
      }

      // Elementwise LTC ODE update (fp32)
#pragma unroll
      for (int j = 0; j < 2; ++j)
#pragma unroll
        for (int v = 0; v < 8; ++v) {
          float act = 1.0f / (1.0f + __expf(-acc[j][v]));
          float gt  = gl[j] + act;
          float ew  = (gl[j] * el[j] + act * er[j]) / (gt + EPSF);
          float hv  = h[j][v];
          h[j][v]   = hv + gt * (ew - hv) * itau[j];
        }
    }

    // Emit h for this timestep
#pragma unroll
    for (int j = 0; j < 2; ++j)
#pragma unroll
      for (int v = 0; v < 8; ++v)
        out[((b0 + v + hi * 8) * TT + t) * HH + ncol[j]] = h[j][v];
  }
}

// ---------------------------------------------------------------------------
// Launcher. Inputs (setup_inputs order):
//   0:x (B,T,I) 1:W_in_w (H,I) 2:W_in_b (H) 3:W_rec_w (H,H)
//   4:tau (H) 5:E_rev (H) 6:gleak (H) 7:Eleak (H)
// Workspace: [0,512K) W_rec f16 | [512K,1M) W_in f16 | [1M,65M) inp_proj f32
// ---------------------------------------------------------------------------
extern "C" void kernel_launch(void* const* d_in, const int* in_sizes, int n_in,
                              void* d_out, int out_size, void* d_ws, size_t ws_size,
                              hipStream_t stream) {
  const float* x      = (const float*)d_in[0];
  const float* W_in_w = (const float*)d_in[1];
  const float* W_in_b = (const float*)d_in[2];
  const float* W_rec  = (const float*)d_in[3];
  const float* tau    = (const float*)d_in[4];
  const float* E_rev  = (const float*)d_in[5];
  const float* gleak  = (const float*)d_in[6];
  const float* Eleak  = (const float*)d_in[7];
  float* out = (float*)d_out;

  char* ws = (char*)d_ws;
  _Float16* wrec16 = (_Float16*)(ws);
  _Float16* win16  = (_Float16*)(ws + (size_t)512 * 1024);
  float*    ipbuf  = (float*)(ws + (size_t)1024 * 1024);

  cvt_f32_f16<<<(HH * HH + 255) / 256, 256, 0, stream>>>(W_rec, wrec16, HH * HH);
  cvt_f32_f16<<<(HH * II + 255) / 256, 256, 0, stream>>>(W_in_w, win16, HH * II);

  // (B*T/16) M-tiles * (H/16) N-tiles = 65536 wave-tiles, 8 waves per block
  input_gemm<<<(BB * TT / 16) * (HH / 16) / 8, 256, 0, stream>>>(x, win16, W_in_b,
                                                                 ipbuf);

  ltc_scan<<<BB / 16, 512, 0, stream>>>(ipbuf, wrec16, tau, E_rev, gleak, Eleak,
                                        out);
}
